// MAPTA_23914377904862
// MI455X (gfx1250) — matrix operand order repaired and checked by reading, not compiled, not run on hardware
//
#include <hip/hip_runtime.h>

#define USE_ASYNC_LDS 1

typedef _Float16 v16h __attribute__((ext_vector_type(16)));
typedef _Float16 v8h  __attribute__((ext_vector_type(8)));
typedef float    v8f  __attribute__((ext_vector_type(8)));
typedef long long ll;

#define TWO_PI_OVER_1024 6.1359231515e-3f

// ---------------- pack z (B,T,N,D) -> zf (BN,T,D) f32 + f16 ----------------
__global__ void k_pack_zf(const float* __restrict__ z, float* __restrict__ zf,
                          _Float16* __restrict__ zfh) {
  ll idx = (ll)blockIdx.x * blockDim.x + threadIdx.x;
  if (idx >= (ll)256 * 1024 * 128) return;
  int d = (int)(idx & 127);
  ll r = idx >> 7;
  int t = (int)(r & 1023);
  ll r2 = r >> 10;
  int n = (int)(r2 & 31);
  int b = (int)(r2 >> 5);
  ll src = (((ll)b * 1024 + t) * 32 + n) * 128 + d;
  float v = z[src];
  zf[idx] = v;
  zfh[idx] = (_Float16)v;
}

// ---------------- DFT basis tables (Fpad=528 x T=1024) in f16 ----------------
__global__ void k_basis(_Float16* __restrict__ cosB, _Float16* __restrict__ sinB) {
  ll idx = (ll)blockIdx.x * blockDim.x + threadIdx.x;
  if (idx >= (ll)528 * 1024) return;
  int t = (int)(idx & 1023);
  int f = (int)(idx >> 10);
  if (f >= 513) { cosB[idx] = (_Float16)0.f; sinB[idx] = (_Float16)0.f; return; }
  int ft = (f * t) & 1023;
  float th = (float)ft * TWO_PI_OVER_1024;
  float s, c;
  __sincosf(th, &s, &c);
  cosB[idx] = (_Float16)c;
  sinB[idx] = (_Float16)s;
}

// ---------------- generic batched WMMA GEMM, LDS-staged 128x64 macro-tiles ----------------
// Fast path: fully in-bounds blocks, double-buffered async global->LDS, 1 barrier/iter.
// Slow path: guarded cooperative loads (edges, B-transposed), single buffer.
// mode 0: f32 store  1: f16 store  2: f32 += (opt f16 mirror)  3: conv-gate accumulate
struct GemmP {
  const _Float16* A; ll lda, bsA;
  const _Float16* B; ll ldb, bsB;
  float* Cf; _Float16* Ch; ll ldc, bsC;
  int M, N, K, bT, mode;
  const float* bias;
  float* outAcc; const float* gate; const float* convB;
  int Np, p, j, pi;
};

__global__ __launch_bounds__(256) void k_wmma_gemm(GemmP P) {
  __shared__ _Float16 lA[2][128 * 32];  // [buf][row 0..127][k 0..31]
  __shared__ _Float16 lB[2][32 * 64];   // [buf][k 0..31][n 0..63]
  const int tid = threadIdx.x;
  const int lane = tid & 31;
  const int wave = tid >> 5;
  const int wm = wave & 3;   // M sub-tile (32 rows each)
  const int wn = wave >> 2;  // N sub-tile (32 cols each)
  const int tileN0 = blockIdx.x * 64;
  const int tileM0 = blockIdx.y * 128;
  const int bz = blockIdx.z;
  const _Float16* A = P.A + (ll)bz * P.bsA;
  const _Float16* Bm = P.B + (ll)bz * P.bsB;
  const int fm = lane & 15;
  const int kb = (lane & 16) ? 8 : 0;

  // per-thread load-chunk coordinates (8 halves each)
  const int aRow = tid >> 2;  // 0..63 (second chunk: +64)
  const int aSeg = tid & 3;
  const int bRow = tid >> 3;  // 0..31
  const int bSeg = tid & 7;

  v8f acc[2][2] = {};

  const bool fast = (tileM0 + 128 <= P.M) && (tileN0 + 64 <= P.N) &&
                    ((P.K & 31) == 0) && ((P.lda & 7) == 0) &&
                    ((P.ldb & 7) == 0) && (!P.bT);

  if (fast) {
    auto issue = [&](int k0, int bufi) {
      const _Float16* ga0 = A + (ll)(tileM0 + aRow) * P.lda + k0 + aSeg * 8;
      const _Float16* ga1 = A + (ll)(tileM0 + aRow + 64) * P.lda + k0 + aSeg * 8;
      const _Float16* gb  = Bm + (ll)(k0 + bRow) * P.ldb + tileN0 + bSeg * 8;
      _Float16* la0 = &lA[bufi][aRow * 32 + aSeg * 8];
      _Float16* la1 = &lA[bufi][(aRow + 64) * 32 + aSeg * 8];
      _Float16* lb  = &lB[bufi][bRow * 64 + bSeg * 8];
#if USE_ASYNC_LDS
      unsigned o0 = (unsigned)(size_t)la0;
      unsigned o1 = (unsigned)(size_t)la1;
      unsigned o2 = (unsigned)(size_t)lb;
      asm volatile("global_load_async_to_lds_b128 %0, %1, off" ::"v"(o0), "v"(ga0) : "memory");
      asm volatile("global_load_async_to_lds_b128 %0, %1, off" ::"v"(o1), "v"(ga1) : "memory");
      asm volatile("global_load_async_to_lds_b128 %0, %1, off" ::"v"(o2), "v"(gb) : "memory");
#else
      *(v8h*)la0 = *(const v8h*)ga0;
      *(v8h*)la1 = *(const v8h*)ga1;
      *(v8h*)lb  = *(const v8h*)gb;
#endif
      __builtin_prefetch(ga0 + 64, 0, 1);
    };
    issue(0, 0);
#if USE_ASYNC_LDS
    asm volatile("s_wait_asynccnt 0x0" ::: "memory");
#endif
    __syncthreads();
    const int nk = P.K >> 5;
    for (int kt = 0; kt < nk; ++kt) {
      const int bufi = kt & 1;
      if (kt + 1 < nk) issue((kt + 1) << 5, 1 - bufi);
      v16h af[2], bfr[2];
#pragma unroll
      for (int ms = 0; ms < 2; ++ms) {
        const _Float16* ap = &lA[bufi][(wm * 32 + ms * 16 + fm) * 32];
        v8h lo = *(const v8h*)(ap + kb);
        v8h hi = *(const v8h*)(ap + 16 + kb);
#pragma unroll
        for (int e = 0; e < 8; ++e) { af[ms][e] = lo[e]; af[ms][e + 8] = hi[e]; }
      }
#pragma unroll
      for (int ns = 0; ns < 2; ++ns) {
        const _Float16* bp = &lB[bufi][wn * 32 + ns * 16 + fm];
#pragma unroll
        for (int e = 0; e < 8; ++e) { bfr[ns][e] = bp[(kb + e) * 64]; bfr[ns][e + 8] = bp[(16 + kb + e) * 64]; }
      }
#pragma unroll
      for (int ms = 0; ms < 2; ++ms)
#pragma unroll
        for (int ns = 0; ns < 2; ++ns)
          acc[ms][ns] = __builtin_amdgcn_wmma_f32_16x16x32_f16(
              false, af[ms], false, bfr[ns], (short)0, acc[ms][ns], false, false);
#if USE_ASYNC_LDS
      asm volatile("s_wait_asynccnt 0x0" ::: "memory");
#endif
      __syncthreads();
    }
  } else {
    const bool aVec = ((P.lda & 7) == 0);
    const bool bVec = (!P.bT) && ((P.ldb & 7) == 0);
    for (int k0 = 0; k0 < P.K; k0 += 32) {
#pragma unroll
      for (int it = 0; it < 2; ++it) {
        int row = aRow + it * 64;
        int gm = tileM0 + row;
        int gk = k0 + aSeg * 8;
        v8h val = {};
        if (gm < P.M) {
          const _Float16* src = A + (ll)gm * P.lda + gk;
          if (aVec && (gk + 8 <= P.K)) {
            val = *(const v8h*)src;
          } else {
#pragma unroll
            for (int e = 0; e < 8; ++e) val[e] = (gk + e < P.K) ? src[e] : (_Float16)0.f;
          }
        }
        *(v8h*)(&lA[0][row * 32 + aSeg * 8]) = val;
      }
      {
        int gk = k0 + bRow;
        int gn = tileN0 + bSeg * 8;
        v8h val = {};
        if (gk < P.K) {
          if (!P.bT) {
            const _Float16* src = Bm + (ll)gk * P.ldb + gn;
            if (bVec && (gn + 8 <= P.N)) {
              val = *(const v8h*)src;
            } else {
#pragma unroll
              for (int e = 0; e < 8; ++e) val[e] = (gn + e < P.N) ? src[e] : (_Float16)0.f;
            }
          } else {
#pragma unroll
            for (int e = 0; e < 8; ++e) {
              int n = gn + e;
              val[e] = (n < P.N) ? Bm[(ll)n * P.ldb + gk] : (_Float16)0.f;
            }
          }
        }
        *(v8h*)(&lB[0][bRow * 64 + bSeg * 8]) = val;
      }
      __syncthreads();
      v16h af[2], bfr[2];
#pragma unroll
      for (int ms = 0; ms < 2; ++ms) {
        const _Float16* ap = &lA[0][(wm * 32 + ms * 16 + fm) * 32];
        v8h lo = *(const v8h*)(ap + kb);
        v8h hi = *(const v8h*)(ap + 16 + kb);
#pragma unroll
        for (int e = 0; e < 8; ++e) { af[ms][e] = lo[e]; af[ms][e + 8] = hi[e]; }
      }
#pragma unroll
      for (int ns = 0; ns < 2; ++ns) {
        const _Float16* bp = &lB[0][wn * 32 + ns * 16 + fm];
#pragma unroll
        for (int e = 0; e < 8; ++e) { bfr[ns][e] = bp[(kb + e) * 64]; bfr[ns][e + 8] = bp[(16 + kb + e) * 64]; }
      }
#pragma unroll
      for (int ms = 0; ms < 2; ++ms)
#pragma unroll
        for (int ns = 0; ns < 2; ++ns)
          acc[ms][ns] = __builtin_amdgcn_wmma_f32_16x16x32_f16(
              false, af[ms], false, bfr[ns], (short)0, acc[ms][ns], false, false);
      __syncthreads();
    }
  }

  // ---- epilogue ----
#pragma unroll
  for (int ms = 0; ms < 2; ++ms) {
#pragma unroll
    for (int ns = 0; ns < 2; ++ns) {
#pragma unroll
      for (int r = 0; r < 8; ++r) {
        int m = tileM0 + wm * 32 + ms * 16 + r + ((lane & 16) ? 8 : 0);
        int n = tileN0 + wn * 32 + ns * 16 + (lane & 15);
        if (m >= P.M || n >= P.N) continue;
        float v = acc[ms][ns][r];
        if (P.bias) v += P.bias[n];
        ll ci = (ll)bz * P.bsC + (ll)m * P.ldc + n;
        if (P.mode == 0) {
          P.Cf[ci] = v;
        } else if (P.mode == 1) {
          P.Ch[ci] = (_Float16)v;
        } else if (P.mode == 2) {
          float v2 = P.Cf[ci] + v;
          P.Cf[ci] = v2;
          if (P.Ch) P.Ch[ci] = (_Float16)v2;  // fused f16 mirror for next GEMM
        } else {
          int bn = m / P.Np;
          int nn = m - bn * P.Np;
          int t = nn * P.p + P.j;
          if (t < 1024) {
            float g = P.gate[((ll)bn * 1024 + t) * 4 + P.pi];
            P.outAcc[(ll)bn * 131072 + (ll)n * 1024 + t] += (v + P.convB[n]) * g;
          }
        }
      }
    }
  }
}

// ---------------- pick top-1 frequency per (bn,d); rank-1 irfft coefficients ----------------
__global__ void k_pick_freq(const float* __restrict__ reC, const float* __restrict__ siC,
                            float* __restrict__ cA, float* __restrict__ cB,
                            int* __restrict__ fstar) {
  ll id = (ll)blockIdx.x * blockDim.x + threadIdx.x;
  if (id >= (ll)256 * 128) return;
  int d = (int)(id & 127);
  int bn = (int)(id >> 7);
  const float* rp = reC + (ll)bn * 528 * 128 + d;
  const float* sp = siC + (ll)bn * 528 * 128 + d;
  float best = -1.f; int bfq = 0; float bre = 0.f, bim = 0.f;
  for (int f = 0; f <= 512; ++f) {
    float re = rp[(ll)f * 128];
    float im = -sp[(ll)f * 128];
    float m2 = re * re + im * im;
    if (m2 > best) { best = m2; bfq = f; bre = re; bim = im; }
  }
  float mag = sqrtf(best);
  float factor = mag / (mag + 1e-8f);
  float w = (bfq == 0 || bfq == 512) ? 1.f : 2.f;
  float c = factor * w * (1.f / 1024.f);
  cA[id] = c * bre;
  cB[id] = c * bim;
  fstar[id] = bfq;
}

// ---------------- seasonal reconstruction + remainder ----------------
__global__ void k_season(const float* __restrict__ zf, const float* __restrict__ cA,
                         const float* __restrict__ cB, const int* __restrict__ fstar,
                         float* __restrict__ z_sea, float* __restrict__ z_rem) {
  ll idx = (ll)blockIdx.x * blockDim.x + threadIdx.x;
  if (idx >= (ll)256 * 1024 * 128) return;
  int d = (int)(idx & 127);
  ll bt = idx >> 7;
  int t = (int)(bt & 1023);
  int bn = (int)(bt >> 10);
  int id = bn * 128 + d;
  int f = fstar[id];
  int ft = (f * t) & 1023;
  float th = (float)ft * TWO_PI_OVER_1024;
  float s, c;
  __sincosf(th, &s, &c);
  float zs = cA[id] * c - cB[id] * s;
  z_sea[idx] = zs;
  z_rem[idx] = zf[idx] - zs;
}

// ---------------- trend gate alpha = softmax(mean_t(z_rem) @ Wq + bq) ----------------
__global__ __launch_bounds__(128) void k_alpha(const float* __restrict__ z_rem,
                                               const float* __restrict__ Wq,
                                               const float* __restrict__ bq,
                                               float* __restrict__ alpha) {
  int bn = blockIdx.x;
  int d = threadIdx.x;
  const float* base = z_rem + (ll)bn * 1024 * 128 + d;
  float s = 0.f;
  for (int t = 0; t < 1024; ++t) s += base[(ll)t * 128];
  __shared__ float mean[128];
  __shared__ float lg[3];
  mean[d] = s * (1.f / 1024.f);
  __syncthreads();
  if (d < 3) {
    float l = bq[d];
    for (int i = 0; i < 128; ++i) l += mean[i] * Wq[i * 3 + d];
    lg[d] = l;
  }
  __syncthreads();
  if (d == 0) {
    float mx = fmaxf(lg[0], fmaxf(lg[1], lg[2]));
    float e0 = expf(lg[0] - mx), e1 = expf(lg[1] - mx), e2 = expf(lg[2] - mx);
    float inv = 1.f / (e0 + e1 + e2);
    alpha[bn * 3 + 0] = e0 * inv;
    alpha[bn * 3 + 1] = e1 * inv;
    alpha[bn * 3 + 2] = e2 * inv;
  }
}

// ---------------- trend (avg pools 3/5/7 mixed by alpha) + f16 concat [zf|sea|trend] ----------------
__global__ void k_trend_cat(const float* __restrict__ zf, const float* __restrict__ z_sea,
                            const float* __restrict__ z_rem, const float* __restrict__ alpha,
                            _Float16* __restrict__ zcat) {
  ll idx = (ll)blockIdx.x * blockDim.x + threadIdx.x;
  if (idx >= (ll)256 * 1024 * 128) return;
  int d = (int)(idx & 127);
  ll bt = idx >> 7;
  int t = (int)(bt & 1023);
  int bn = (int)(bt >> 10);
  const float* zr = z_rem + ((ll)bn * 1024) * 128 + d;
  float w[7];
#pragma unroll
  for (int j = 0; j < 7; ++j) {
    int tt = t + j - 3;
    w[j] = (tt >= 0 && tt < 1024) ? zr[(ll)tt * 128] : 0.f;
  }
  float a3 = w[2] + w[3] + w[4];
  float a5 = a3 + w[1] + w[5];
  float a7 = a5 + w[0] + w[6];
  const float* al = alpha + bn * 3;
  float trend = al[0] * a3 * (1.f / 3.f) + al[1] * a5 * (1.f / 5.f) + al[2] * a7 * (1.f / 7.f);
  _Float16* row = zcat + bt * 384;
  row[d] = (_Float16)zf[idx];
  row[128 + d] = (_Float16)z_sea[idx];
  row[256 + d] = (_Float16)trend;
}

// ---------------- noisy top-3 router (wave per (bn,t) row) ----------------
__global__ __launch_bounds__(128) void k_router(const float* __restrict__ z_fused,
                                                const float* __restrict__ noise_feat,
                                                const float* __restrict__ noise_logits,
                                                const float* __restrict__ Wr,
                                                const float* __restrict__ br,
                                                const float* __restrict__ Wn,
                                                const float* __restrict__ bn_,
                                                float* __restrict__ sparse) {
  int lane = threadIdx.x & 31;
  int wave = threadIdx.x >> 5;
  ll row = (ll)blockIdx.x * 4 + wave;
  if (row >= (ll)256 * 1024) return;
  const float* zr = z_fused + row * 128;
  const float* nf = noise_feat + row * 128;
  float aR[4] = {}, aN[4] = {};
#pragma unroll
  for (int u = 0; u < 4; ++u) {
    int d = lane + 32 * u;
    float g = zr[d] + 0.1f * nf[d];
#pragma unroll
    for (int k = 0; k < 4; ++k) { aR[k] += g * Wr[d * 4 + k]; aN[k] += g * Wn[d * 4 + k]; }
  }
#pragma unroll
  for (int k = 0; k < 4; ++k) {
    for (int off = 16; off > 0; off >>= 1) {
      aR[k] += __shfl_xor(aR[k], off, 32);
      aN[k] += __shfl_xor(aN[k], off, 32);
    }
  }
  if (lane == 0) {
    float l[4], e[4];
    float mx = -1e30f;
    for (int k = 0; k < 4; ++k) {
      float snv = aN[k] + bn_[k];
      float sp = (snv > 20.f) ? snv : log1pf(expf(snv));
      l[k] = aR[k] + br[k] + noise_logits[row * 4 + k] * sp;
      mx = fmaxf(mx, l[k]);
    }
    float se = 0.f;
    for (int k = 0; k < 4; ++k) { e[k] = expf(l[k] - mx); se += e[k]; }
    float inv = 1.f / se;
    int mn = 0; float mv = e[0];
    for (int k = 1; k < 4; ++k) if (e[k] < mv) { mv = e[k]; mn = k; }
    for (int k = 0; k < 4; ++k) sparse[row * 4 + k] = (k == mn) ? 0.f : e[k] * inv;
  }
}

// ---------------- patch helpers ----------------
__global__ void k_fill_patches(const _Float16* __restrict__ zfh, _Float16* __restrict__ ph,
                               int Tp) {
  ll idx = (ll)blockIdx.x * blockDim.x + threadIdx.x;
  if (idx >= (ll)256 * Tp * 128) return;
  int d = (int)(idx & 127);
  ll m = idx >> 7;
  int t = (int)(m % Tp);
  int bn = (int)(m / Tp);
  ph[idx] = (t < 1024) ? zfh[((ll)bn * 1024 + t) * 128 + d] : (_Float16)0.f;
}

__global__ void k_qmean(const float* __restrict__ zf, _Float16* __restrict__ qm, int Np, int p) {
  ll idx = (ll)blockIdx.x * blockDim.x + threadIdx.x;
  if (idx >= (ll)256 * Np * 128) return;
  int d = (int)(idx & 127);
  ll m = idx >> 7;
  int n = (int)(m % Np);
  int bn = (int)(m / Np);
  float s = 0.f;
  for (int j = 0; j < p; ++j) {
    int t = n * p + j;
    if (t < 1024) s += zf[((ll)bn * 1024 + t) * 128 + d];
  }
  qm[idx] = (_Float16)(s / (float)p);
}

// ---------------- local patch attention (wave per (bn,patch)) ----------------
__global__ __launch_bounds__(128) void k_local_attn(const _Float16* __restrict__ qh,
                                                    const _Float16* __restrict__ kkh,
                                                    const _Float16* __restrict__ vvh,
                                                    float* __restrict__ AO, int Np, int p,
                                                    float scale) {
  int lane = threadIdx.x & 31;
  int wave = threadIdx.x >> 5;
  ll row = (ll)blockIdx.x * 4 + wave;
  if (row >= (ll)256 * Np) return;
  const _Float16* qrow = qh + row * 128;
  float qv[4];
#pragma unroll
  for (int u = 0; u < 4; ++u) qv[u] = (float)qrow[lane + 32 * u];
  float lg[8];
  for (int j = 0; j < p; ++j) {
    const _Float16* krow = kkh + (row * p + j) * 128;
    float dd = 0.f;
#pragma unroll
    for (int u = 0; u < 4; ++u) dd += qv[u] * (float)krow[lane + 32 * u];
    for (int off = 16; off > 0; off >>= 1) dd += __shfl_xor(dd, off, 32);
    lg[j] = dd * scale;
  }
  float mx = lg[0];
  for (int j = 1; j < p; ++j) mx = fmaxf(mx, lg[j]);
  float se = 0.f;
  for (int j = 0; j < p; ++j) { lg[j] = expf(lg[j] - mx); se += lg[j]; }
  float inv = 1.f / se;
  float ov[4] = {};
  for (int j = 0; j < p; ++j) {
    float wj = lg[j] * inv;
    const _Float16* vrow = vvh + (row * p + j) * 128;
#pragma unroll
    for (int u = 0; u < 4; ++u) ov[u] += wj * (float)vrow[lane + 32 * u];
  }
  float* orow = AO + row * 128;
#pragma unroll
  for (int u = 0; u < 4; ++u) orow[lane + 32 * u] = ov[u];
}

// ---------------- row-wise softmax(sigmoid(S)*scale) -> f16 probs ----------------
__global__ __launch_bounds__(256) void k_sigsoftmax(const float* __restrict__ S,
                                                    _Float16* __restrict__ Pr, int Np,
                                                    float scale) {
  ll row = blockIdx.x;
  const float* src = S + row * (ll)Np;
  _Float16* dst = Pr + row * (ll)Np;
  int tid = threadIdx.x;
  __shared__ float sh[256];
  float lv[2];
  int cnt = 0;
  float mx = -1e30f;
  for (int n = tid; n < Np; n += 256) {
    float v = scale / (1.f + expf(-src[n]));
    lv[cnt++] = v;
    mx = fmaxf(mx, v);
  }
  sh[tid] = mx; __syncthreads();
  for (int s = 128; s > 0; s >>= 1) { if (tid < s) sh[tid] = fmaxf(sh[tid], sh[tid + s]); __syncthreads(); }
  mx = sh[0]; __syncthreads();
  float sum = 0.f;
  cnt = 0;
  for (int n = tid; n < Np; n += 256) { lv[cnt] = expf(lv[cnt] - mx); sum += lv[cnt]; ++cnt; }
  sh[tid] = sum; __syncthreads();
  for (int s = 128; s > 0; s >>= 1) { if (tid < s) sh[tid] += sh[tid + s]; __syncthreads(); }
  float inv = 1.f / sh[0];
  cnt = 0;
  for (int n = tid; n < Np; n += 256) dst[n] = (_Float16)(lv[cnt++] * inv);
}

// ---------------- misc ----------------
__global__ void k_f32_to_f16(const float* __restrict__ s, _Float16* __restrict__ d, ll n) {
  ll i = (ll)blockIdx.x * blockDim.x + threadIdx.x;
  if (i < n) d[i] = (_Float16)s[i];
}

__global__ void k_pack_conv(const float* __restrict__ src, _Float16* __restrict__ dst, int p) {
  ll idx = (ll)blockIdx.x * blockDim.x + threadIdx.x;
  if (idx >= (ll)p * 128 * 128) return;
  int j = (int)(idx / 16384);
  int rem = (int)(idx - (ll)j * 16384);
  int i = rem >> 7;
  int o = rem & 127;
  dst[idx] = (_Float16)src[(ll)(i * 128 + o) * p + j];
}

__global__ void k_zero(float* __restrict__ p, ll n) {
  ll i = (ll)blockIdx.x * blockDim.x + threadIdx.x;
  if (i < n) p[i] = 0.f;
}

// dout[b,t,n,d] = outAcc_flat[(b*32+n)*131072 + t*128 + d]  (replicates reference reshape quirk)
__global__ void k_final(const float* __restrict__ outAcc, float* __restrict__ dout) {
  ll idx = (ll)blockIdx.x * blockDim.x + threadIdx.x;
  if (idx >= (ll)256 * 1024 * 128) return;
  int d = (int)(idx & 127);
  ll r = idx >> 7;
  int n = (int)(r & 31);
  ll r2 = r >> 5;
  int t = (int)(r2 & 1023);
  int b = (int)(r2 >> 10);
  ll src = ((ll)(b * 32 + n)) * 131072 + (ll)t * 128 + d;
  dout[idx] = outAcc[src];
}

extern "C" void kernel_launch(void* const* d_in, const int* in_sizes, int n_in,
                              void* d_out, int out_size, void* d_ws, size_t ws_size,
                              hipStream_t stream) {
  (void)in_sizes; (void)n_in; (void)out_size; (void)ws_size;
  const float* z            = (const float*)d_in[0];
  const float* noise_feat   = (const float*)d_in[1];
  const float* noise_logits = (const float*)d_in[2];
  const float* Wq_tr = (const float*)d_in[3];
  const float* bq_tr = (const float*)d_in[4];
  const float* Wf    = (const float*)d_in[5];
  const float* bf    = (const float*)d_in[6];
  const float* Wr    = (const float*)d_in[7];
  const float* br    = (const float*)d_in[8];
  const float* Wn    = (const float*)d_in[9];
  const float* bnb   = (const float*)d_in[10];
  const float* lqkv_w = (const float*)d_in[11];
  const float* lqkv_b = (const float*)d_in[12];
  const float* gqkv_b = (const float*)d_in[13];
  const float* proj_w = (const float*)d_in[14];
  const float* proj_b = (const float*)d_in[15];
  const float* conv_b = (const float*)d_in[16];
  const float* gqkv_w[4] = {(const float*)d_in[17], (const float*)d_in[19],
                            (const float*)d_in[21], (const float*)d_in[23]};
  const float* conv_w[4] = {(const float*)d_in[18], (const float*)d_in[20],
                            (const float*)d_in[22], (const float*)d_in[24]};
  float* out = (float*)d_out;

  size_t off = 0;
  auto alloc = [&](size_t bytes) -> char* {
    char* p = (char*)d_ws + off;
    off += (bytes + 255) & ~(size_t)255;
    return p;
  };

  const ll TOT = (ll)256 * 1024 * 128;
  float*    zf    = (float*)alloc(TOT * 4);
  _Float16* zfh   = (_Float16*)alloc(TOT * 2);
  _Float16* cosB  = (_Float16*)alloc((ll)528 * 1024 * 2);
  _Float16* sinB  = (_Float16*)alloc((ll)528 * 1024 * 2);
  float*    reC   = (float*)alloc((ll)256 * 528 * 128 * 4);
  float*    siC   = (float*)alloc((ll)256 * 528 * 128 * 4);
  float*    cA    = (float*)alloc((ll)256 * 128 * 4);
  float*    cBc   = (float*)alloc((ll)256 * 128 * 4);
  int*      fst   = (int*)alloc((ll)256 * 128 * 4);
  float*    z_sea = (float*)alloc(TOT * 4);
  float*    z_rem = (float*)alloc(TOT * 4);
  float*    alphaB = (float*)alloc(256 * 3 * 4);
  _Float16* zcat  = (_Float16*)alloc((ll)256 * 1024 * 384 * 2);
  float*    z_fused = (float*)alloc(TOT * 4);
  float*    sparse  = (float*)alloc((ll)256 * 1024 * 4 * 4);
  float*    outAcc  = (float*)alloc(TOT * 4);
  _Float16* WfH   = (_Float16*)alloc((ll)49152 * 2);
  _Float16* lqkvH = (_Float16*)alloc((ll)196608 * 2);
  _Float16* projH = (_Float16*)alloc((ll)65536 * 2);
  _Float16* gqkvH = (_Float16*)alloc((ll)983040 * 2);
  _Float16* convH = (_Float16*)alloc((ll)327680 * 2);
  const int NpMax = 512, TpMax = 1026;
  _Float16* patchesH = (_Float16*)alloc((ll)256 * TpMax * 128 * 2);
  _Float16* qmH   = (_Float16*)alloc((ll)256 * NpMax * 128 * 2);
  _Float16* qH    = (_Float16*)alloc((ll)256 * NpMax * 128 * 2);
  _Float16* kkH   = (_Float16*)alloc((ll)256 * TpMax * 128 * 2);
  _Float16* vvH   = (_Float16*)alloc((ll)256 * TpMax * 128 * 2);
  float*    AO    = (float*)alloc((ll)256 * NpMax * 128 * 4);
  _Float16* AOh   = (_Float16*)alloc((ll)256 * NpMax * 128 * 2);
  _Float16* gqH   = (_Float16*)alloc((ll)256 * NpMax * 128 * 2);
  _Float16* gkH   = (_Float16*)alloc((ll)256 * NpMax * 128 * 2);
  _Float16* gvH   = (_Float16*)alloc((ll)256 * NpMax * 128 * 2);
  float*    Sbuf  = (float*)alloc((ll)256 * NpMax * NpMax * 4);
  _Float16* PrH   = (_Float16*)alloc((ll)256 * NpMax * NpMax * 2);
  _Float16* fusedH = (_Float16*)alloc((ll)256 * NpMax * 128 * 2);

  auto gemm = [&](const _Float16* A, ll lda, ll bsA, const _Float16* Bp, ll ldb, ll bsB,
                  int bT, float* Cf, _Float16* Ch, ll ldc, ll bsC, int M, int Nn, int K,
                  int batch, int mode, const float* bias, float* oAcc, const float* gate,
                  const float* cvB, int Np_, int p_, int j_, int pi_) {
    GemmP P{A, lda, bsA, Bp, ldb, bsB, Cf, Ch, ldc, bsC,
            M, Nn, K, bT, mode, bias, oAcc, gate, cvB, Np_, p_, j_, pi_};
    dim3 grid((unsigned)((Nn + 63) / 64), (unsigned)((M + 127) / 128), (unsigned)batch);
    k_wmma_gemm<<<grid, 256, 0, stream>>>(P);
  };

  // Stage 1: pack + DFT basis
  k_pack_zf<<<(unsigned)(TOT / 256), 256, 0, stream>>>(z, zf, zfh);
  k_basis<<<(528 * 1024) / 256, 256, 0, stream>>>(cosB, sinB);

  // Stage 2: batched DFT via WMMA (528x1024 basis @ 1024x128 series), then top-1 freq
  gemm(cosB, 1024, 0, zfh, 128, (ll)1024 * 128, 0, reC, nullptr, 128, (ll)528 * 128,
       528, 128, 1024, 256, 0, nullptr, nullptr, nullptr, nullptr, 0, 0, 0, 0);
  gemm(sinB, 1024, 0, zfh, 128, (ll)1024 * 128, 0, siC, nullptr, 128, (ll)528 * 128,
       528, 128, 1024, 256, 0, nullptr, nullptr, nullptr, nullptr, 0, 0, 0, 0);
  k_pick_freq<<<(256 * 128) / 256, 256, 0, stream>>>(reC, siC, cA, cBc, fst);
  k_season<<<(unsigned)(TOT / 256), 256, 0, stream>>>(zf, cA, cBc, fst, z_sea, z_rem);

  // Stage 3: trend
  k_alpha<<<256, 128, 0, stream>>>(z_rem, Wq_tr, bq_tr, alphaB);
  k_trend_cat<<<(unsigned)(TOT / 256), 256, 0, stream>>>(zf, z_sea, z_rem, alphaB, zcat);

  // Weight conversions
  k_f32_to_f16<<<(49152 + 255) / 256, 256, 0, stream>>>(Wf, WfH, 49152);
  k_f32_to_f16<<<(196608 + 255) / 256, 256, 0, stream>>>(lqkv_w, lqkvH, 196608);
  k_f32_to_f16<<<(65536 + 255) / 256, 256, 0, stream>>>(proj_w, projH, 65536);
  const int ps[4] = {2, 4, 6, 8};
  const int Nps[4] = {512, 256, 171, 128};
  const ll go[4] = {0, 98304, 294912, 589824};
  const ll co[4] = {0, 32768, 98304, 196608};
  for (int i = 0; i < 4; ++i) {
    ll cnt = (ll)3 * ps[i] * 128 * 128;
    k_f32_to_f16<<<(unsigned)((cnt + 255) / 256), 256, 0, stream>>>(gqkv_w[i], gqkvH + go[i], cnt);
    ll cc = (ll)ps[i] * 128 * 128;
    k_pack_conv<<<(unsigned)((cc + 255) / 256), 256, 0, stream>>>(conv_w[i], convH + co[i], ps[i]);
  }

  // Stage 4: fusion GEMM + router
  gemm(zcat, 384, 0, WfH, 128, 0, 0, z_fused, nullptr, 128, 0, 262144, 128, 384, 1, 0, bf,
       nullptr, nullptr, nullptr, 0, 0, 0, 0);
  k_router<<<65536, 128, 0, stream>>>(z_fused, noise_feat, noise_logits, Wr, br, Wn, bnb, sparse);
  k_zero<<<(unsigned)(TOT / 256), 256, 0, stream>>>(outAcc, TOT);

  // Stage 5: per-patch branches
  const float scale = 0.08838834764831845f;  // 128^-0.5
  for (int i = 0; i < 4; ++i) {
    int p = ps[i], Np = Nps[i], Tp = Np * p;
    int Mq = 256 * Np;
    int Mk = 256 * Tp;
    const _Float16* patP = zfh;
    if (Tp != 1024) {
      k_fill_patches<<<(unsigned)(((ll)256 * Tp * 128 + 255) / 256), 256, 0, stream>>>(zfh, patchesH, Tp);
      patP = patchesH;
    }
    k_qmean<<<(unsigned)(((ll)Mq * 128 + 255) / 256), 256, 0, stream>>>(zf, qmH, Np, p);
    gemm(qmH, 128, 0, lqkvH + (ll)(i * 3 + 0) * 16384, 128, 0, 0, nullptr, qH, 128, 0,
         Mq, 128, 128, 1, 1, lqkv_b + (i * 3 + 0) * 128, nullptr, nullptr, nullptr, 0, 0, 0, 0);
    gemm(patP, 128, 0, lqkvH + (ll)(i * 3 + 1) * 16384, 128, 0, 0, nullptr, kkH, 128, 0,
         Mk, 128, 128, 1, 1, lqkv_b + (i * 3 + 1) * 128, nullptr, nullptr, nullptr, 0, 0, 0, 0);
    gemm(patP, 128, 0, lqkvH + (ll)(i * 3 + 2) * 16384, 128, 0, 0, nullptr, vvH, 128, 0,
         Mk, 128, 128, 1, 1, lqkv_b + (i * 3 + 2) * 128, nullptr, nullptr, nullptr, 0, 0, 0, 0);
    k_local_attn<<<(unsigned)((Mq + 3) / 4), 128, 0, stream>>>(qH, kkH, vvH, AO, Np, p, scale);
    _Float16* gts[3] = {gqH, gkH, gvH};
    for (int c = 0; c < 3; ++c)
      gemm(patP, (ll)p * 128, 0, gqkvH + go[i] + (ll)c * p * 128 * 128, 128, 0, 0,
           nullptr, gts[c], 128, 0, Mq, 128, p * 128, 1, 1, gqkv_b + (i * 3 + c) * 128,
           nullptr, nullptr, nullptr, 0, 0, 0, 0);
    gemm(gqH, 128, (ll)Np * 128, gkH, 128, (ll)Np * 128, 1, Sbuf, nullptr, Np, (ll)Np * Np,
         Np, Np, 128, 256, 0, nullptr, nullptr, nullptr, nullptr, 0, 0, 0, 0);
    k_sigsoftmax<<<(unsigned)(256 * Np), 256, 0, stream>>>(Sbuf, PrH, Np, scale);
    // global attention output: AO += Pr @ gv, with fused f16 mirror into AOh
    gemm(PrH, Np, (ll)Np * Np, gvH, 128, (ll)Np * 128, 0, AO, AOh, 128, (ll)Np * 128,
         Np, 128, Np, 256, 2, nullptr, nullptr, nullptr, nullptr, 0, 0, 0, 0);
    gemm(AOh, 128, 0, projH + (ll)i * 16384, 128, 0, 0, nullptr, fusedH, 128, 0,
         Mq, 128, 128, 1, 1, proj_b + i * 128, nullptr, nullptr, nullptr, 0, 0, 0, 0);
    for (int j = 0; j < p; ++j)
      gemm(fusedH, 128, 0, convH + co[i] + (ll)j * 16384, 128, 0, 0, nullptr, nullptr, 0, 0,
           Mq, 128, 128, 1, 3, nullptr, outAcc, sparse, conv_b + i * 128, Np, p, j, i);
  }

  k_final<<<(unsigned)(TOT / 256), 256, 0, stream>>>(outAcc, out);
}
